// RayFeatureHead_67946382623322
// MI455X (gfx1250) — matrix operand order repaired and checked by reading, not compiled, not run on hardware
//
#include <hip/hip_runtime.h>
#include <hip/hip_bf16.h>

// ---------------- problem constants (fixed by reference) ----------------
#define Bn   2
#define CIN  32
#define CB   16
#define CH   32
#define Dd   96
#define HV   96
#define WV   96
#define HP   256
#define WP   256
#define NSTEP 17                 // (8+8)/1 + 1
#define DHW  (Dd*HV*WV)          // 884736
#define NPIX (HP*WP)             // 65536
#define NRAY (Bn*NPIX)           // 131072
#define NEG_FILL (-3.402823466e38f)

typedef _Float16 v16h __attribute__((ext_vector_type(16)));
typedef _Float16 half8 __attribute__((ext_vector_type(8)));
typedef float    v8f  __attribute__((ext_vector_type(8)));

__device__ inline v8f wmma_f16(const v16h& a, const v16h& b, const v8f& c) {
    return __builtin_amdgcn_wmma_f32_16x16x32_f16(
        /*neg_a=*/false, a, /*neg_b=*/false, b,
        /*c_mod=*/(short)0, c, /*reuse_a=*/false, /*reuse_b=*/false);
}

// Branch-free erf (Abramowitz & Stegun 7.1.26, |err| <= 1.5e-7) -> exact-GELU.
// Uses v_rcp_f32 / v_exp_f32 directly: no EXEC-divergent regions around WMMA.
__device__ inline float gelu_exact(float x) {
    float u = x * 0.70710678118654752f;       // x / sqrt(2)
    float a = fabsf(u);
    float t = __builtin_amdgcn_rcpf(fmaf(0.3275911f, a, 1.0f));
    float p =             1.061405429f;
    p = fmaf(p, t, -1.453152027f);
    p = fmaf(p, t,  1.421413741f);
    p = fmaf(p, t, -0.284496736f);
    p = fmaf(p, t,  0.254829592f);
    p = p * t;
    float e  = __builtin_amdgcn_exp2f(-a * a * 1.44269504088896f); // exp(-a^2)
    float er = 1.0f - p * e;                  // erf(|u|)
    er = copysignf(er, u);
    return 0.5f * x * (1.0f + er);
}

// WMMA 16-bit A-matrix (16x32) half-index -> K mapping (contiguous: K = 16*(lane/16) + i, matches B fills)
__device__ inline int a_kidx(int lane, int i) {
    return (lane < 16) ? i
                       : i + 16;
}

// ======================================================================
// Kernel 1: project (B,CIN,D,H,W) f32 -> (B,D,H,W,CB) f16   (mem-bound)
// Channel-last f16 output (56 MB) fits in the 192 MB L2 for stage 2.
// ======================================================================
__global__ void __launch_bounds__(256)
proj_kernel(const float* __restrict__ vol,
            const float* __restrict__ pw,   // (CB, CIN)
            const float* __restrict__ pb,   // (CB,)
            _Float16*    __restrict__ dst)  // (B, D*H*W, CB) f16
{
    long idx = (long)blockIdx.x * 256 + threadIdx.x;   // over B*DHW voxels
    if (idx >= (long)Bn * DHW) return;
    int b = (int)(idx / DHW);
    int v = (int)(idx % DHW);
    const float* src = vol + (long)b * CIN * DHW + v;

    float acc[CB];
#pragma unroll
    for (int o = 0; o < CB; ++o) acc[o] = pb[o];
#pragma unroll
    for (int c = 0; c < CIN; ++c) {
        float x = src[(long)c * DHW];
#pragma unroll
        for (int o = 0; o < CB; ++o) acc[o] += pw[o * CIN + c] * x;
    }
    half8 lo, hi;
#pragma unroll
    for (int o = 0; o < 8; ++o) { lo[o] = (_Float16)acc[o]; hi[o] = (_Float16)acc[8 + o]; }
    half8* out = (half8*)(dst + idx * CB);
    out[0] = lo; out[1] = hi;
}

// ======================================================================
// Kernel 2: sample + WMMA MLP + masked max-pool.
// 64 threads = 2 waves; each wave owns 16 rays (N=16 WMMA columns).
// ======================================================================
__global__ void __launch_bounds__(64)
ray_kernel(const _Float16* __restrict__ proj,   // (B, DHW, CB) f16
           const float* __restrict__ pts,       // (B,HP,WP,3)
           const float* __restrict__ nrm,       // (B,HP,WP,3)
           const unsigned char* __restrict__ vmask, // (B,HP,WP) bool
           const float* __restrict__ w1, const float* __restrict__ b1, // (CH,CB,3),(CH)
           const float* __restrict__ w2, const float* __restrict__ b2, // (CH,CH,3),(CH)
           const float* __restrict__ w3, const float* __restrict__ b3, // (1,CH,1),(1)
           float* __restrict__ out)             // [NRAY pooled | NRAY valid]
{
    const int tid    = threadIdx.x;
    const int w      = tid >> 5;        // wave id in block (0..1)
    const int lane   = tid & 31;
    const int ray    = lane & 15;       // ray owned for B/D fragments
    const bool lo16  = (lane < 16);

    // X: sampled features, padded steps [0]=zero, [1..17]=steps, [18]=zero
    __shared__ v16h Xs[2][16][NSTEP + 2];
    // h1: conv1 activations, 32 ch = 2 x v16h, same step padding
    __shared__ v16h H1s[2][16][NSTEP + 2][2];
    __shared__ unsigned char SVs[2][16][NSTEP];

    // ---- zero the padding rows ----
    if (lo16) {
        v16h z;
#pragma unroll
        for (int i = 0; i < 16; ++i) z[i] = (_Float16)0.f;
        Xs[w][lane][0] = z;  Xs[w][lane][NSTEP + 1] = z;
        H1s[w][lane][0][0] = z;          H1s[w][lane][0][1] = z;
        H1s[w][lane][NSTEP + 1][0] = z;  H1s[w][lane][NSTEP + 1][1] = z;
    }

    const long gbase = (long)blockIdx.x * 32 + (long)w * 16;  // first ray gid of wave

    // =========== Stage A: trilinear sampling (17 steps x 16 rays) ===========
    for (int idx = lane; idx < 16 * NSTEP; idx += 32) {
        int r = idx / NSTEP;
        int s = idx % NSTEP;
        long gid = gbase + r;
        int  b   = (int)(gid / NPIX);
        int  pix = (int)(gid % NPIX);
        const float* pp = pts + ((long)b * NPIX + pix) * 3;
        const float* np = nrm + ((long)b * NPIX + pix) * 3;
        float off = (float)(s - 8);
        float z = pp[0] + np[0] * off;
        float y = pp[1] + np[1] * off;
        float x = pp[2] + np[2] * off;
        bool sv = vmask[gid] != 0;
        sv = sv && (fabsf(z) <= 3.2e38f) && (z >= 0.f) && (z <= (float)(Dd - 1));
        sv = sv && (fabsf(y) <= 3.2e38f) && (y >= 0.f) && (y <= (float)(HV - 1));
        sv = sv && (fabsf(x) <= 3.2e38f) && (x >= 0.f) && (x <= (float)(WV - 1));

        float z0 = floorf(z), y0 = floorf(y), x0 = floorf(x);
        float dz = z - z0, dy = y - y0, dx = x - x0;
        float acc[CB];
#pragma unroll
        for (int c = 0; c < CB; ++c) acc[c] = 0.f;
        const _Float16* vb = proj + (long)b * DHW * CB;
#pragma unroll
        for (int cz = 0; cz < 2; ++cz) {
            float iz = z0 + (float)cz;
            float wz = cz ? dz : 1.f - dz;
            bool okz = (iz >= 0.f) && (iz < (float)Dd);
            int zi = (int)fminf(fmaxf(iz, 0.f), (float)(Dd - 1));
#pragma unroll
            for (int cy = 0; cy < 2; ++cy) {
                float iy = y0 + (float)cy;
                float wy = cy ? dy : 1.f - dy;
                bool oky = (iy >= 0.f) && (iy < (float)HV);
                int yi = (int)fminf(fmaxf(iy, 0.f), (float)(HV - 1));
#pragma unroll
                for (int cx = 0; cx < 2; ++cx) {
                    float ix = x0 + (float)cx;
                    float wx = cx ? dx : 1.f - dx;
                    bool okx = (ix >= 0.f) && (ix < (float)WV);
                    int xi = (int)fminf(fmaxf(ix, 0.f), (float)(WV - 1));
                    float wgt = (okz && oky && okx) ? (wz * wy * wx) : 0.f;
                    const _Float16* cp = vb + (((long)zi * HV + yi) * WV + xi) * CB;
#pragma unroll
                    for (int c = 0; c < CB; ++c) acc[c] += wgt * (float)cp[c];
                }
            }
        }
        half8 hl, hh;
#pragma unroll
        for (int c = 0; c < 8; ++c) { hl[c] = (_Float16)acc[c]; hh[c] = (_Float16)acc[8 + c]; }
        half8* xp = (half8*)&Xs[w][r][s + 1];
        xp[0] = hl; xp[1] = hh;
        SVs[w][r][s] = sv ? 1 : 0;
    }
    __syncthreads();

    // =========== Stage B: conv1 (16->32, k=3) via WMMA f16 16x16x32 ===========
    // im2col K = tap*16 + c (48 valid, 16 zero-pad)  -> 2 K-tiles, 2 M-tiles
    v16h a1[2][2];
#pragma unroll
    for (int mt = 0; mt < 2; ++mt) {
        int oc = mt * 16 + ray;   // A rows: lanes L and L+16 share row L&15
#pragma unroll
        for (int kt = 0; kt < 2; ++kt) {
#pragma unroll
            for (int i = 0; i < 16; ++i) {
                int K = kt * 32 + a_kidx(lane, i);
                float v = 0.f;
                if (K < 48) { int tap = K >> 4; int c = K & 15; v = w1[oc * 48 + c * 3 + tap]; }
                a1[mt][kt][i] = (_Float16)v;
            }
        }
    }
    v8f bias1[2];
#pragma unroll
    for (int mt = 0; mt < 2; ++mt)
#pragma unroll
        for (int r8 = 0; r8 < 8; ++r8)
            bias1[mt][r8] = b1[mt * 16 + (lo16 ? r8 : 8 + r8)];

    for (int t = 0; t < NSTEP; ++t) {
        // K-tile0: lanes0-15 hold K0..15 (tap0 -> padded row t), lanes16-31 K16..31 (tap1 -> t+1)
        v16h bk0 = Xs[w][ray][lo16 ? t : t + 1];
        // K-tile1: lanes0-15 hold K32..47 (tap2 -> t+2), lanes16-31 K48..63 = zero pad row 0
        v16h bk1 = Xs[w][ray][lo16 ? t + 2 : 0];
#pragma unroll
        for (int mt = 0; mt < 2; ++mt) {
            v8f acc = bias1[mt];
            acc = wmma_f16(a1[mt][0], bk0, acc);
            acc = wmma_f16(a1[mt][1], bk1, acc);
            // D rows: lane<16 -> M=r, lane>=16 -> M=8+r ; out ch = mt*16+M
            half8 hv;
#pragma unroll
            for (int r8 = 0; r8 < 8; ++r8) hv[r8] = (_Float16)gelu_exact(acc[r8]);
            _Float16* hp = (_Float16*)&H1s[w][ray][t + 1][0];
            *(half8*)(hp + mt * 16 + (lo16 ? 0 : 8)) = hv;
        }
    }
    __syncthreads();

    // =========== Stage C: conv2 (32->32, k=3) + conv3 (32->1) + pool ===========
    // im2col K = tap*32 + c  -> exactly 3 K-tiles of 32, 2 M-tiles
    v16h a2[2][3];
#pragma unroll
    for (int mt = 0; mt < 2; ++mt) {
        int oc = mt * 16 + ray;
#pragma unroll
        for (int kt = 0; kt < 3; ++kt) {
#pragma unroll
            for (int i = 0; i < 16; ++i) {
                int c = a_kidx(lane, i);         // K within tile = channel
                a2[mt][kt][i] = (_Float16)w2[oc * 96 + c * 3 + kt];
            }
        }
    }
    v8f bias2[2];
    float w3l[16];
#pragma unroll
    for (int mt = 0; mt < 2; ++mt)
#pragma unroll
        for (int r8 = 0; r8 < 8; ++r8) {
            int ch = mt * 16 + (lo16 ? r8 : 8 + r8);
            bias2[mt][r8] = b2[ch];
            w3l[mt * 8 + r8] = w3[ch];
        }

    float runmax = NEG_FILL;
    int   anyv   = 0;
    const int sel = lo16 ? 0 : 1;   // B frag: lane<16 reads ch0..15, lane>=16 ch16..31
    for (int t = 0; t < NSTEP; ++t) {
        v16h h0 = H1s[w][ray][t][sel];
        v16h h1 = H1s[w][ray][t + 1][sel];
        v16h h2 = H1s[w][ray][t + 2][sel];
        float partial = 0.f;
#pragma unroll
        for (int mt = 0; mt < 2; ++mt) {
            v8f acc = bias2[mt];
            acc = wmma_f16(a2[mt][0], h0, acc);
            acc = wmma_f16(a2[mt][1], h1, acc);
            acc = wmma_f16(a2[mt][2], h2, acc);
#pragma unroll
            for (int r8 = 0; r8 < 8; ++r8)
                partial += gelu_exact(acc[r8]) * w3l[mt * 8 + r8];
        }
        // lanes L and L+16 hold complementary channel halves of the same ray
        float full = partial + __shfl_xor(partial, 16, 32) + b3[0];
        int sv = SVs[w][ray][t];
        float logit = sv ? full : NEG_FILL;
        runmax = fmaxf(runmax, logit);
        anyv |= sv;
    }
    if (lo16) {
        long gid = gbase + ray;
        out[gid]        = anyv ? runmax : 0.f;
        out[NRAY + gid] = anyv ? 1.f : 0.f;
    }
}

// ======================================================================
extern "C" void kernel_launch(void* const* d_in, const int* in_sizes, int n_in,
                              void* d_out, int out_size, void* d_ws, size_t ws_size,
                              hipStream_t stream) {
    const float*         vol = (const float*)d_in[0];
    const float*         pts = (const float*)d_in[1];
    const float*         nrm = (const float*)d_in[2];
    const unsigned char* val = (const unsigned char*)d_in[3];
    const float*         pw  = (const float*)d_in[4];
    const float*         pb  = (const float*)d_in[5];
    const float*         w1  = (const float*)d_in[6];
    const float*         b1  = (const float*)d_in[7];
    const float*         w2  = (const float*)d_in[8];
    const float*         b2  = (const float*)d_in[9];
    const float*         w3  = (const float*)d_in[10];
    const float*         b3  = (const float*)d_in[11];
    _Float16* proj = (_Float16*)d_ws;       // needs B*DHW*CB*2 = ~56.6 MB
    float*    out  = (float*)d_out;

    const int nvox_blocks = (Bn * DHW + 255) / 256;      // 6912
    proj_kernel<<<nvox_blocks, 256, 0, stream>>>(vol, pw, pb, proj);

    const int nray_blocks = NRAY / 32;                   // 4096 (2 waves x 16 rays)
    ray_kernel<<<nray_blocks, 64, 0, stream>>>(proj, pts, nrm, val,
                                               w1, b1, w2, b2, w3, b3, out);
}
